// NemotronFlashMamba2_58205396795473
// MI455X (gfx1250) — matrix operand order repaired and checked by reading, not compiled, not run on hardware
//
#include <hip/hip_runtime.h>

typedef __attribute__((ext_vector_type(16))) _Float16 v16h;
typedef __attribute__((ext_vector_type(8)))  _Float16 v8h;
typedef __attribute__((ext_vector_type(8)))  float    v8f;

#define B_SZ      2
#define SEQ       2048
#define D_MODEL   2048
#define D_INNER   4096
#define D_STATE   128
#define HEADS     64
#define PDIM      64
#define CHUNK     128
#define NCHUNK    (SEQ / CHUNK)                       // 16
#define CONV_DIM  (D_INNER + 2 * D_STATE)             // 4352
#define D_IN_PROJ (2 * D_INNER + 2 * D_STATE + HEADS) // 8512
#define ROWS      (B_SZ * SEQ)                        // 4096

// ---------------- helpers ----------------
__device__ __forceinline__ float sigmoidf_(float x) { return 1.0f / (1.0f + __expf(-x)); }
__device__ __forceinline__ float siluf_(float x)    { return x * sigmoidf_(x); }
__device__ __forceinline__ float softplusf_(float x){ return (x > 20.0f) ? x : log1pf(__expf(x)); }

__device__ __forceinline__ v16h make_v16(v8h a, v8h b) {
  return __builtin_shufflevector(a, b, 0,1,2,3,4,5,6,7,8,9,10,11,12,13,14,15);
}
// A-fragment (16x32 f16): lane row = M, halves K = kh0..kh0+7 and kh0+16..kh0+23
__device__ __forceinline__ v16h load_afrag(const _Float16* rowptr, int kh0) {
  v8h a0 = *(const v8h*)(rowptr + kh0);
  v8h a1 = *(const v8h*)(rowptr + kh0 + 16);
  return make_v16(a0, a1);
}
// B-fragment (32x16 f16): lane col = N, halves K = kb..kb+7 and kb+16..kb+23 (kb = 0 or 8)
__device__ __forceinline__ v16h load_bfrag(const _Float16* colptr, int kb0) {
  const int kb = kb0 - (kb0 & 16) + ((kb0 & 16) >> 1);
  v8h b0 = *(const v8h*)(colptr + kb);
  v8h b1 = *(const v8h*)(colptr + kb + 16);
  return make_v16(b0, b1);
}

// ---------------- Tensor Data Mover (CDNA5 TDM) path ----------------
#if defined(__gfx1250__) && __has_builtin(__builtin_amdgcn_tensor_load_to_lds) && \
    __has_builtin(__builtin_amdgcn_s_wait_tensorcnt)
#define USE_TDM 1
#endif

#ifdef USE_TDM
typedef __attribute__((ext_vector_type(4))) unsigned int u32x4;
typedef __attribute__((ext_vector_type(4))) int          i32x4;
typedef __attribute__((ext_vector_type(8))) int          i32x8;

// DMA a [tile_rows x 32] f16 tile (row pitch = stride elems) from global into LDS,
// padding each 64B row with 16B so LDS rows are 40 halfs (matches As/Bs tiles).
__device__ __forceinline__ void tdm_load_tile_f16(const _Float16* gsrc, void* lds_dst,
                                                  int tile_rows, long stride_elems)
{
  unsigned long long ga = (unsigned long long)(size_t)gsrc;
  unsigned int lds = (unsigned int)(size_t)lds_dst;   // LDS aperture: low 32 bits = LDS byte offset
  u32x4 g0;
  g0[0] = 1u;                                              // count = 1 (valid descriptor)
  g0[1] = lds;                                             // lds_addr
  g0[2] = (unsigned int)ga;                                // global_addr[31:0]
  g0[3] = ((unsigned int)(ga >> 32) & 0x01FFFFFFu) | (2u << 30); // addr[56:32] | type=2
  unsigned int td0 = 1u << 24, td1 = 1u << 24;             // generous tensor dims (tile in-bounds)
  unsigned long long s0 = (unsigned long long)stride_elems;
  i32x8 g1;
  g1[0] = (int)((1u << 16)      // data_size = 2 bytes
              | (1u << 20)      // pad_enable
              | (3u << 22)      // pad_interval: every 16 DWORDs (= one 64B row)
              | (3u << 25));    // pad_amount: 4 DWORDs (= 16B -> 40-half LDS rows)
  g1[1] = (int)((td0 & 0xFFFFu) << 16);                    // tensor_dim0[15:0]
  g1[2] = (int)((td0 >> 16) | ((td1 & 0xFFFFu) << 16));    // tensor_dim0[31:16], dim1[15:0]
  g1[3] = (int)((td1 >> 16) | (32u << 16));                // tensor_dim1[31:16], tile_dim0=32
  g1[4] = (int)(tile_rows & 0xFFFF);                       // tile_dim1, tile_dim2=0
  g1[5] = (int)(unsigned int)(s0 & 0xFFFFFFFFu);           // tensor_dim0_stride[31:0]
  g1[6] = (int)(unsigned int)(s0 >> 32);                   // stride[47:32], dim1_stride lo=0
  g1[7] = 0;
  i32x4 z4 = {0, 0, 0, 0};
#if __clang_major__ >= 23
  i32x8 z8 = {0, 0, 0, 0, 0, 0, 0, 0};
  __builtin_amdgcn_tensor_load_to_lds(g0, g1, z4, z4, z8, 0);
#else
  __builtin_amdgcn_tensor_load_to_lds(g0, g1, z4, z4, 0);
#endif
}
#endif // USE_TDM

// ---------------- f32 -> f16 convert ----------------
__global__ __launch_bounds__(256)
void k_cvt_f16(const float* __restrict__ in, _Float16* __restrict__ out, int n) {
  int i = blockIdx.x * 256 + threadIdx.x;
  if (i < n) out[i] = (_Float16)in[i];
}

// ---------------- generic NT GEMM: C[m,n] = sum_k A[m,k]*B[n,k] ----------------
// block tile 128(M) x 64(N), BK=32, 256 threads (8 wave32s), batching via blockIdx.z.
// TDM path: double-buffered LDS tiles; wave 0 streams tile k+1 via tensor_load_to_lds
// while all waves run WMMAs on tile k (s_wait_tensorcnt(2) = previous pair landed).
template <typename OutT>
__global__ __launch_bounds__(256)
void k_gemm_nt(const _Float16* __restrict__ A, const _Float16* __restrict__ Bm,
               OutT* __restrict__ C, int M, int N, int K, int lda, int ldb, int ldc,
               long sA, long sB, long sC)
{
  __shared__ _Float16 As[2][128][40];
  __shared__ _Float16 Bs[2][64][40];
  A  += (long)blockIdx.z * sA;
  Bm += (long)blockIdx.z * sB;
  C  += (long)blockIdx.z * sC;
  const int t = threadIdx.x, wave = t >> 5, lane = t & 31;
  const int m0 = blockIdx.x * 128, n0 = blockIdx.y * 64;
  const int fr = lane & 15;
  const int lrow = wave * 16 + fr;
  const int kh0 = (lane >> 4) * 8;
  const int kb0 = (lane >> 4) * 16;
  v8f acc[4] = {};
  const int nk = K >> 5;
#ifdef USE_TDM
  if (wave == 0) {
    tdm_load_tile_f16(A  + (long)m0 * lda, &As[0][0][0], 128, lda);
    tdm_load_tile_f16(Bm + (long)n0 * ldb, &Bs[0][0][0],  64, ldb);
  }
  for (int i = 0; i < nk; ++i) {
    const int cur = i & 1;
    if (wave == 0) {
      if (i + 1 < nk) {
        int k1 = (i + 1) << 5;
        tdm_load_tile_f16(A  + (long)m0 * lda + k1, &As[cur ^ 1][0][0], 128, lda);
        tdm_load_tile_f16(Bm + (long)n0 * ldb + k1, &Bs[cur ^ 1][0][0],  64, ldb);
        __builtin_amdgcn_s_wait_tensorcnt(2);  // pair for tile i complete; i+1 in flight
      } else {
        __builtin_amdgcn_s_wait_tensorcnt(0);  // final tile complete
      }
    }
    __syncthreads();
    v16h a = load_afrag(&As[cur][lrow][0], kh0);
#pragma unroll
    for (int nt = 0; nt < 4; ++nt) {
      v16h b = load_bfrag(&Bs[cur][nt * 16 + fr][0], kb0);
      acc[nt] = __builtin_amdgcn_wmma_f32_16x16x32_f16(false, a, false, b, (short)0,
                                                       acc[nt], false, false);
    }
    __syncthreads();   // all reads done before next DMA overwrites the other buffer
  }
#else
  for (int i = 0; i < nk; ++i) {
    const int k0 = i << 5;
    {
      const int arow = t >> 1, acol = (t & 1) * 16;
      const int brow = t >> 2, bcol = (t & 3) * 8;
      const _Float16* ap = A + (long)(m0 + arow) * lda + (k0 + acol);
      *(v8h*)&As[0][arow][acol]     = *(const v8h*)ap;
      *(v8h*)&As[0][arow][acol + 8] = *(const v8h*)(ap + 8);
      const _Float16* bp = Bm + (long)(n0 + brow) * ldb + (k0 + bcol);
      *(v8h*)&Bs[0][brow][bcol] = *(const v8h*)bp;
      if (k0 + 32 < K) {           // prefetch next K tiles (global_prefetch_b8)
        __builtin_prefetch(ap + 32, 0, 1);
        __builtin_prefetch(bp + 32, 0, 1);
      }
    }
    __syncthreads();
    v16h a = load_afrag(&As[0][lrow][0], kh0);
#pragma unroll
    for (int nt = 0; nt < 4; ++nt) {
      v16h b = load_bfrag(&Bs[0][nt * 16 + fr][0], kb0);
      acc[nt] = __builtin_amdgcn_wmma_f32_16x16x32_f16(false, a, false, b, (short)0,
                                                       acc[nt], false, false);
    }
    __syncthreads();
  }
#endif
#pragma unroll
  for (int nt = 0; nt < 4; ++nt)
#pragma unroll
    for (int r = 0; r < 8; ++r) {
      int row = m0 + wave * 16 + r + 8 * (lane >> 4);
      int col = n0 + nt * 16 + fr;
      C[(long)row * ldc + col] = (OutT)acc[nt][r];
    }
}

// ---------------- depthwise causal conv (width 4) + bias + SiLU ----------------
__global__ __launch_bounds__(256)
void k_conv_silu(const float* __restrict__ zxbcdt, const float* __restrict__ Wc,
                 const float* __restrict__ bconv, float* __restrict__ xBC,
                 _Float16* __restrict__ B16, _Float16* __restrict__ C16)
{
  int idx = blockIdx.x * 256 + threadIdx.x;
  if (idx >= ROWS * CONV_DIM) return;
  int ch = idx % CONV_DIM;
  int rg = idx / CONV_DIM;
  int tt = rg % SEQ;
  int b  = rg / SEQ;
  float s = bconv[ch];
#pragma unroll
  for (int j = 0; j < 4; ++j) {
    int tj = tt - 3 + j;
    if (tj >= 0)
      s += zxbcdt[(long)(b * SEQ + tj) * D_IN_PROJ + D_INNER + ch] * Wc[ch * 4 + j];
  }
  float v = siluf_(s);
  xBC[(long)rg * CONV_DIM + ch] = v;
  if (ch >= D_INNER) {
    int n = ch - D_INNER;
    if (n < D_STATE) B16[rg * D_STATE + n] = (_Float16)v;
    else             C16[rg * D_STATE + (n - D_STATE)] = (_Float16)v;
  }
}

// ---------------- softplus(dt+bias), dA = dt*A, per-chunk inclusive cumsum ----------------
__global__ __launch_bounds__(128)
void k_dt_scan(const float* __restrict__ zxbcdt, const float* __restrict__ dt_bias,
               const float* __restrict__ A_log, float* __restrict__ dt_post,
               float* __restrict__ dAcs)
{
  __shared__ float s[CHUNK];
  int bc = blockIdx.x >> 6;  // b*NCHUNK + c
  int h  = blockIdx.x & 63;
  int l  = threadIdx.x;
  int row = bc * CHUNK + l;
  float raw = zxbcdt[(long)row * D_IN_PROJ + (D_IN_PROJ - HEADS) + h] + dt_bias[h];
  float dtv = softplusf_(raw);
  dt_post[row * HEADS + h] = dtv;
  s[l] = -__expf(A_log[h]) * dtv;
  __syncthreads();
  for (int off = 1; off < CHUNK; off <<= 1) {
    float v = (l >= off) ? s[l - off] : 0.0f;
    __syncthreads();
    s[l] += v;
    __syncthreads();
  }
  dAcs[blockIdx.x * CHUNK + l] = s[l];
}

// ---------------- per (b,chunk,head): Y_diag = M@u , states = Bd^T@u ----------------
// MB buffer is used twice: first holds M (masked-decay matrix), then re-staged as Bd^T.
__global__ __launch_bounds__(256)
void k_chunk_core(const float* __restrict__ xBC, const float* __restrict__ dt_post,
                  const float* __restrict__ dAcs, const _Float16* __restrict__ CB16,
                  float* __restrict__ Ybuf, float* __restrict__ states)
{
  __shared__ float    sdA[CHUNK];
  __shared__ _Float16 uT[PDIM][CHUNK + 8];
  __shared__ _Float16 MB[D_STATE][CHUNK + 8];
  const int bch = blockIdx.x;          // (b*NCHUNK+c)*HEADS + h
  const int bc = bch >> 6, h = bch & 63;
  const int rowBase = bc * CHUNK;
  const int t = threadIdx.x;
  if (t < CHUNK) sdA[t] = dAcs[bch * CHUNK + t];
  __syncthreads();
  const float dA_last = sdA[CHUNK - 1];
  // stage u^T: u[l,p] = x[l, h*64+p] * dt[l,h]
#pragma unroll 4
  for (int i = 0; i < 32; ++i) {
    int idx = t + 256 * i;
    int l = idx >> 6, p = idx & 63;
    int row = rowBase + l;
    float u = xBC[(long)row * CONV_DIM + h * PDIM + p] * dt_post[row * HEADS + h];
    uT[p][l] = (_Float16)u;
  }
  // stage M[l][s] = CB[l,s] * exp(dAcs[l]-dAcs[s]) * (s<=l)  (bulk, WMMA stays divergence-free)
#pragma unroll 4
  for (int i = 0; i < 64; ++i) {
    int idx = t + 256 * i;
    int l = idx >> 7, ss = idx & 127;
    float cb = (float)CB16[(long)(bc * CHUNK + l) * CHUNK + ss];
    float e = __expf(sdA[l] - sdA[ss]);
    float msk = (ss <= l) ? 1.0f : 0.0f;
    MB[l][ss] = (_Float16)(cb * e * msk);
  }
  __syncthreads();
  const int wave = t >> 5, lane = t & 31;
  const int fr = lane & 15;
  const int kh0 = (lane >> 4) * 8;
  const int kb0 = (lane >> 4) * 16;
  const int lrow = wave * 16 + fr;
  // ---- Y_diag = M @ u
  {
    v8f acc[4] = {};
#pragma unroll
    for (int ki = 0; ki < 4; ++ki) {
      v16h a = load_afrag(&MB[lrow][0], ki * 32 + kh0);
#pragma unroll
      for (int nt = 0; nt < 4; ++nt) {
        v16h b = load_bfrag(&uT[nt * 16 + fr][0], ki * 32 + kb0);
        acc[nt] = __builtin_amdgcn_wmma_f32_16x16x32_f16(false, a, false, b, (short)0,
                                                         acc[nt], false, false);
      }
    }
#pragma unroll
    for (int nt = 0; nt < 4; ++nt)
#pragma unroll
      for (int r = 0; r < 8; ++r) {
        int row = rowBase + wave * 16 + r + 8 * (lane >> 4);
        int col = h * PDIM + nt * 16 + fr;
        Ybuf[(long)row * D_INNER + col] = acc[nt][r];
      }
  }
  __syncthreads();
  // re-stage MB as Bd^T: Bd[l,n] = B[l,n]*exp(dA_last - dAcs[l]), stored [n][l]
#pragma unroll 4
  for (int i = 0; i < 64; ++i) {
    int idx = t + 256 * i;
    int l = idx >> 7, n = idx & 127;
    int row = rowBase + l;
    float decay = __expf(dA_last - sdA[l]);
    MB[n][l] = (_Float16)(xBC[(long)row * CONV_DIM + D_INNER + n] * decay);
  }
  __syncthreads();
  // ---- states[n,p] = sum_l Bd[l,n]*u[l,p]
  {
    v8f acc[4] = {};
#pragma unroll
    for (int ki = 0; ki < 4; ++ki) {
      v16h a = load_afrag(&MB[lrow][0], ki * 32 + kh0);
#pragma unroll
      for (int nt = 0; nt < 4; ++nt) {
        v16h b = load_bfrag(&uT[nt * 16 + fr][0], ki * 32 + kb0);
        acc[nt] = __builtin_amdgcn_wmma_f32_16x16x32_f16(false, a, false, b, (short)0,
                                                         acc[nt], false, false);
      }
    }
#pragma unroll
    for (int nt = 0; nt < 4; ++nt)
#pragma unroll
      for (int r = 0; r < 8; ++r) {
        int n = wave * 16 + r + 8 * (lane >> 4);
        int p = nt * 16 + fr;
        states[((long)bch * D_STATE + n) * PDIM + p] = acc[nt][r];
      }
  }
}

// ---------------- inter-chunk recurrent scan over NCHUNK ----------------
__global__ __launch_bounds__(256)
void k_state_scan(const float* __restrict__ dAcs, const float* __restrict__ states,
                  float* __restrict__ hstarts)
{
  int b = blockIdx.x >> 6, h = blockIdx.x & 63;
  int t = threadIdx.x;
  float hreg[32];
#pragma unroll
  for (int i = 0; i < 32; ++i) hreg[i] = 0.0f;
  for (int c = 0; c < NCHUNK; ++c) {
    int bch = (b * NCHUNK + c) * HEADS + h;
    float cd = __expf(dAcs[bch * CHUNK + CHUNK - 1]);
    long base = (long)bch * (D_STATE * PDIM);
#pragma unroll 4
    for (int i = 0; i < 32; ++i) {
      int e = t + 256 * i;
      hstarts[base + e] = hreg[i];
      hreg[i] = hreg[i] * cd + states[base + e];
    }
  }
}

// ---------------- Y_off = (C * exp(dAcs)) @ h_starts ; Y += Y_off + D*x ----------------
__global__ __launch_bounds__(256)
void k_y_off(const float* __restrict__ xBC, const float* __restrict__ dAcs,
             const _Float16* __restrict__ C16, const float* __restrict__ hstarts,
             const float* __restrict__ Dvec, float* __restrict__ Ybuf)
{
  __shared__ float    sdA[CHUNK];
  __shared__ _Float16 hT[PDIM][D_STATE + 8];
  const int bch = blockIdx.x;
  const int bc = bch >> 6, h = bch & 63;
  const int rowBase = bc * CHUNK;
  const int t = threadIdx.x;
  if (t < CHUNK) sdA[t] = dAcs[bch * CHUNK + t];
  long hbase = (long)bch * (D_STATE * PDIM);
#pragma unroll 4
  for (int i = 0; i < 32; ++i) {
    int idx = t + 256 * i;
    int n = idx >> 6, p = idx & 63;
    hT[p][n] = (_Float16)hstarts[hbase + n * PDIM + p];
  }
  __syncthreads();
  const int wave = t >> 5, lane = t & 31;
  const int fr = lane & 15;
  const int kh0 = (lane >> 4) * 8;
  const int kb0 = (lane >> 4) * 16;
  const int lrow = wave * 16 + fr;
  const float sdo = __expf(sdA[lrow]);
  const float Dh = Dvec[h];
  v8f acc[4] = {};
  const _Float16* crow = C16 + (long)(rowBase + lrow) * D_STATE;
#pragma unroll
  for (int ki = 0; ki < 4; ++ki) {
    int s0 = ki * 32 + kh0;
    v8h c0 = *(const v8h*)(crow + s0);
    v8h c1 = *(const v8h*)(crow + s0 + 16);
    v16h a;
#pragma unroll
    for (int j = 0; j < 8; ++j) {
      a[j]     = (_Float16)((float)c0[j] * sdo);
      a[8 + j] = (_Float16)((float)c1[j] * sdo);
    }
#pragma unroll
    for (int nt = 0; nt < 4; ++nt) {
      v16h b = load_bfrag(&hT[nt * 16 + fr][0], ki * 32 + kb0);
      acc[nt] = __builtin_amdgcn_wmma_f32_16x16x32_f16(false, a, false, b, (short)0,
                                                       acc[nt], false, false);
    }
  }
#pragma unroll
  for (int nt = 0; nt < 4; ++nt)
#pragma unroll
    for (int r = 0; r < 8; ++r) {
      int row = rowBase + wave * 16 + r + 8 * (lane >> 4);
      int col = h * PDIM + nt * 16 + fr;
      float xv = xBC[(long)row * CONV_DIM + col];
      long yi = (long)row * D_INNER + col;
      Ybuf[yi] = Ybuf[yi] + acc[nt][r] + Dh * xv;
    }
}

// ---------------- gate with silu(z) + RMSNorm -> f16 ----------------
__global__ __launch_bounds__(256)
void k_gate_rmsnorm(const float* __restrict__ Ybuf, const float* __restrict__ zxbcdt,
                    const float* __restrict__ nw, _Float16* __restrict__ y16)
{
  __shared__ float red[256];
  int row = blockIdx.x, t = threadIdx.x;
  float g[16];
  float acc = 0.0f;
#pragma unroll
  for (int i = 0; i < 16; ++i) {
    int col = t + 256 * i;
    float z = zxbcdt[(long)row * D_IN_PROJ + col];
    float gv = Ybuf[(long)row * D_INNER + col] * siluf_(z);
    g[i] = gv;
    acc += gv * gv;
  }
  red[t] = acc;
  __syncthreads();
  for (int o = 128; o > 0; o >>= 1) {
    if (t < o) red[t] += red[t + o];
    __syncthreads();
  }
  float scale = rsqrtf(red[0] / (float)D_INNER + 1e-5f);
#pragma unroll
  for (int i = 0; i < 16; ++i) {
    int col = t + 256 * i;
    y16[(long)row * D_INNER + col] = (_Float16)(g[i] * scale * nw[col]);
  }
}

// ---------------- host launcher ----------------
extern "C" void kernel_launch(void* const* d_in, const int* in_sizes, int n_in,
                              void* d_out, int out_size, void* d_ws, size_t ws_size,
                              hipStream_t stream)
{
  (void)in_sizes; (void)n_in; (void)out_size; (void)ws_size;
  const float* hidden  = (const float*)d_in[0];
  const float* W_in    = (const float*)d_in[1];
  const float* W_conv  = (const float*)d_in[2];
  const float* b_conv  = (const float*)d_in[3];
  const float* A_log   = (const float*)d_in[4];
  const float* Dvec    = (const float*)d_in[5];
  const float* dt_bias = (const float*)d_in[6];
  const float* norm_w  = (const float*)d_in[7];
  const float* W_out   = (const float*)d_in[8];
  float* out = (float*)d_out;

  char* p = (char*)d_ws;
  auto carve = [&](size_t bytes) -> char* {
    char* r = p;
    p += (bytes + 255) & ~(size_t)255;
    return r;
  };
  float*    zxbcdt  = (float*)   carve((size_t)ROWS * D_IN_PROJ * 4);
  float*    xBC     = (float*)   carve((size_t)ROWS * CONV_DIM * 4);
  float*    dt_post = (float*)   carve((size_t)ROWS * HEADS * 4);
  float*    dAcs    = (float*)   carve((size_t)B_SZ * NCHUNK * HEADS * CHUNK * 4);
  _Float16* B16     = (_Float16*)carve((size_t)ROWS * D_STATE * 2);
  _Float16* C16     = (_Float16*)carve((size_t)ROWS * D_STATE * 2);
  _Float16* CB16    = (_Float16*)carve((size_t)B_SZ * NCHUNK * CHUNK * CHUNK * 2);
  float*    Ybuf    = (float*)   carve((size_t)ROWS * D_INNER * 4);
  float*    states  = (float*)   carve((size_t)B_SZ * NCHUNK * HEADS * D_STATE * PDIM * 4);
  float*    hstart  = (float*)   carve((size_t)B_SZ * NCHUNK * HEADS * D_STATE * PDIM * 4);
  _Float16* y16     = (_Float16*)carve((size_t)ROWS * D_INNER * 2);
  _Float16* hs16    = (_Float16*)carve((size_t)ROWS * D_MODEL * 2);
  _Float16* Win16   = (_Float16*)carve((size_t)D_IN_PROJ * D_MODEL * 2);
  _Float16* Wout16  = (_Float16*)carve((size_t)D_MODEL * D_INNER * 2);

  int n1 = ROWS * D_MODEL;
  k_cvt_f16<<<(n1 + 255) / 256, 256, 0, stream>>>(hidden, hs16, n1);
  int n2 = D_IN_PROJ * D_MODEL;
  k_cvt_f16<<<(n2 + 255) / 256, 256, 0, stream>>>(W_in, Win16, n2);
  int n3 = D_MODEL * D_INNER;
  k_cvt_f16<<<(n3 + 255) / 256, 256, 0, stream>>>(W_out, Wout16, n3);

  // in_proj: zxbcdt[4096, 8512]
  k_gemm_nt<float><<<dim3(ROWS / 128, D_IN_PROJ / 64, 1), 256, 0, stream>>>(
      hs16, Win16, zxbcdt, ROWS, D_IN_PROJ, D_MODEL, D_MODEL, D_MODEL, D_IN_PROJ,
      0L, 0L, 0L);

  int nc = ROWS * CONV_DIM;
  k_conv_silu<<<(nc + 255) / 256, 256, 0, stream>>>(zxbcdt, W_conv, b_conv, xBC, B16, C16);

  k_dt_scan<<<B_SZ * NCHUNK * HEADS, CHUNK, 0, stream>>>(zxbcdt, dt_bias, A_log,
                                                         dt_post, dAcs);

  // CB[l,s] = sum_n C[l,n]*B[s,n], batched per (b,chunk)
  k_gemm_nt<_Float16><<<dim3(1, 2, B_SZ * NCHUNK), 256, 0, stream>>>(
      C16, B16, CB16, CHUNK, CHUNK, D_STATE, D_STATE, D_STATE, CHUNK,
      (long)CHUNK * D_STATE, (long)CHUNK * D_STATE, (long)CHUNK * CHUNK);

  k_chunk_core<<<B_SZ * NCHUNK * HEADS, 256, 0, stream>>>(xBC, dt_post, dAcs, CB16,
                                                          Ybuf, states);

  k_state_scan<<<B_SZ * HEADS, 256, 0, stream>>>(dAcs, states, hstart);

  k_y_off<<<B_SZ * NCHUNK * HEADS, 256, 0, stream>>>(xBC, dAcs, C16, hstart, Dvec, Ybuf);

  k_gate_rmsnorm<<<ROWS, 256, 0, stream>>>(Ybuf, zxbcdt, norm_w, y16);

  // out_proj: out[4096, 2048]
  k_gemm_nt<float><<<dim3(ROWS / 128, D_MODEL / 64, 1), 256, 0, stream>>>(
      y16, Wout16, out, ROWS, D_MODEL, D_INNER, D_INNER, D_INNER, D_MODEL,
      0L, 0L, 0L);
}
